// PyGRandLANet_21406117003723
// MI455X (gfx1250) — matrix operand order repaired and checked by reading, not compiled, not run on hardware
//
#include <hip/hip_runtime.h>
#include <hip/hip_bf16.h>

#define LRELU_SLOPE 0.2f
#define BN_EPS 1e-6f

typedef __attribute__((ext_vector_type(16))) _Float16 v16h;
typedef __attribute__((ext_vector_type(8)))  float    v8f;

// ---------------------------------------------------------------------------
// Weight packing: W(KxN, f32, row-major) -> per-lane f16 B-fragments.
// Fragment layout (16-bit B 32x16): lane<16 -> col=tile*16+lane, k=e;
// lane>=16 -> col=tile*16+lane-16, k=16+e.  v16h element e=2d+h lives in
// dword d (lo/hi half).  Storage: dword index = ((tn*KC+kc)*32 + lane)*8 + d.
// All bounds logic lives here; pads are zero so the GEMM needs no guards.
// ---------------------------------------------------------------------------
__global__ __launch_bounds__(32) void k_packw(const float* __restrict__ W,
                                              unsigned int* __restrict__ Wp,
                                              int K, int N, int KC) {
  const int lane = threadIdx.x & 31;
  const int tn = blockIdx.x / KC;
  const int kc = blockIdx.x % KC;
  const int half = lane >> 4, l16 = lane & 15;
  const int colN = tn * 16 + l16;
  unsigned int out[8];
#pragma unroll
  for (int e = 0; e < 16; e += 2) {
    const int k = kc * 32 + half * 16 + e;
    _Float16 lo = (colN < N && k     < K) ? (_Float16)W[(size_t)k * N + colN]       : (_Float16)0.0f;
    _Float16 hi = (colN < N && (k+1) < K) ? (_Float16)W[(size_t)(k + 1) * N + colN] : (_Float16)0.0f;
    union { _Float16 h[2]; unsigned int u; } p;
    p.h[0] = lo; p.h[1] = hi;
    out[e >> 1] = p.u;
  }
  unsigned int* dst = Wp + ((size_t)blockIdx.x * 32 + lane) * 8;
#pragma unroll
  for (int d = 0; d < 8; ++d) dst[d] = out[d];
}

// ---------------------------------------------------------------------------
// Fast WMMA GEMM: C[MxN] = A[MxK] @ W (+bias), one wave32 per 16x16 tile.
// Preconditions handled by host: M % 16 == 0; A resides in workspace with
// >=128B tail slack (K-tail overread multiplies zero-padded B -> exact).
// Inner loop: unconditional b32 A loads, b128 packed-B loads, cvt, wmma.
// ---------------------------------------------------------------------------
__device__ __forceinline__ v8f gemm_chunk(const float* __restrict__ Ar,
                                          const unsigned int* __restrict__ Wt,
                                          int kc, int half, v8f acc) {
  const int k0 = kc * 32;
  v16h a;
#pragma unroll
  for (int e = 0; e < 8; ++e) a[e] = (_Float16)Ar[k0 + 16 * half + e];
#pragma unroll
  for (int e = 0; e < 8; ++e) a[8 + e] = (_Float16)Ar[k0 + 16 * half + 8 + e];
  union { uint4 q[2]; v16h h; } bu;
  const unsigned int* wp = Wt + (size_t)kc * 32 * 8;
  bu.q[0] = *(const uint4*)(wp);
  bu.q[1] = *(const uint4*)(wp + 4);
  return __builtin_amdgcn_wmma_f32_16x16x32_f16(false, a, false, bu.h,
                                                (short)0, acc, false, false);
}

__global__ __launch_bounds__(32) void k_gemm_f(const float* __restrict__ A,
                                               const unsigned int* __restrict__ Wp,
                                               const float* __restrict__ bias,
                                               float* __restrict__ C,
                                               int M, int K, int N, int KC) {
  const int lane = threadIdx.x & 31;
  const int half = lane >> 4, l16 = lane & 15;
  const int row  = blockIdx.x * 16 + l16;   // always < M (M % 16 == 0)
  const int colN = blockIdx.y * 16 + l16;
  const bool colOK = (colN < N);

  const float* Ar = A + (size_t)row * K;
  const unsigned int* Wt = Wp + (((size_t)blockIdx.y * KC) * 32 + lane) * 8;

  float bb = 0.0f;
  if (bias != nullptr) bb = bias[colOK ? colN : 0];

  v8f acc = {};
  int kc = 0;
  for (; kc + 2 <= KC; kc += 2) {
    __builtin_prefetch(Ar + kc * 32 + 64, 0, 3);
    acc = gemm_chunk(Ar, Wt, kc, half, acc);
    acc = gemm_chunk(Ar, Wt, kc + 1, half, acc);
  }
  if (kc < KC) acc = gemm_chunk(Ar, Wt, kc, half, acc);

  // D layout: VGPR v -> row = 16*tileM + v + 8*half, col = 16*tileN + l16
  if (colOK) {
#pragma unroll
    for (int v = 0; v < 8; ++v) {
      const int r = blockIdx.x * 16 + v + 8 * half;
      C[(size_t)r * N + colN] = acc[v] + bb;
    }
  }
}

// ------------------- per-column batch-norm statistics ----------------------
__global__ void k_colstats(const float* __restrict__ X, int M, int N,
                           float* __restrict__ mu, float* __restrict__ var) {
  const int c = blockIdx.x;
  __shared__ float ss[256];
  __shared__ float sq[256];
  float s = 0.0f, q = 0.0f;
  for (int r = threadIdx.x; r < M; r += blockDim.x) {
    float v = X[(size_t)r * N + c];
    s += v; q += v * v;
  }
  ss[threadIdx.x] = s; sq[threadIdx.x] = q;
  __syncthreads();
  for (int off = 128; off > 0; off >>= 1) {
    if ((int)threadIdx.x < off) {
      ss[threadIdx.x] += ss[threadIdx.x + off];
      sq[threadIdx.x] += sq[threadIdx.x + off];
    }
    __syncthreads();
  }
  if (threadIdx.x == 0) {
    float m = ss[0] / (float)M;
    mu[c]  = m;
    var[c] = sq[0] / (float)M - m * m;
  }
}

// ------------------- apply BN (+ optional LeakyReLU), in place -------------
__global__ void k_bn_act(float* __restrict__ Y, const float* __restrict__ mu,
                         const float* __restrict__ var, const float* __restrict__ g,
                         const float* __restrict__ be, int total, int N, int act) {
  int t = blockIdx.x * blockDim.x + threadIdx.x;
  if (t >= total) return;
  int c = t % N;
  float v = (Y[t] - mu[c]) * rsqrtf(var[c] + BN_EPS) * g[c] + be[c];
  if (act) v = (v >= 0.0f) ? v : LRELU_SLOPE * v;
  Y[t] = v;
}

// ------------------- brute-force KNN (top-16 smallest d2) ------------------
__global__ void k_knn(const float* __restrict__ posQ, const float* __restrict__ posA,
                      int m, int n, int* __restrict__ col) {
  int q = blockIdx.x * blockDim.x + threadIdx.x;
  if (q >= m) return;
  const float qx = posQ[3 * q + 0], qy = posQ[3 * q + 1], qz = posQ[3 * q + 2];
  float bd[16]; int bi[16];
#pragma unroll
  for (int k = 0; k < 16; ++k) { bd[k] = 3.4e38f; bi[k] = 0; }
  for (int p = 0; p < n; ++p) {
    if ((p & 127) == 0 && p + 128 < n) __builtin_prefetch(&posA[3 * (p + 128)], 0, 3);
    float dx = posA[3 * p + 0] - qx;
    float dy = posA[3 * p + 1] - qy;
    float dz = posA[3 * p + 2] - qz;
    float d2 = dx * dx + dy * dy + dz * dz;
    if (d2 < bd[15]) {
      int j = 15;
      while (j > 0 && bd[j - 1] > d2) { bd[j] = bd[j - 1]; bi[j] = bi[j - 1]; --j; }
      bd[j] = d2; bi[j] = p;
    }
  }
  for (int k = 0; k < 16; ++k) col[q * 16 + k] = bi[k];
}

// ------------------- 1-NN (argmin over source points) ----------------------
__global__ void k_nn1(const float* __restrict__ posSkip, int Ms,
                      const float* __restrict__ posSrc, int Mx,
                      int* __restrict__ nn) {
  int i = blockIdx.x * blockDim.x + threadIdx.x;
  if (i >= Ms) return;
  const float sx = posSkip[3 * i + 0], sy = posSkip[3 * i + 1], sz = posSkip[3 * i + 2];
  float best = 3.4e38f; int bi = 0;
  for (int p = 0; p < Mx; ++p) {
    float dx = posSrc[3 * p + 0] - sx;
    float dy = posSrc[3 * p + 1] - sy;
    float dz = posSrc[3 * p + 2] - sz;
    float d2 = dx * dx + dy * dy + dz * dz;
    if (d2 < best) { best = d2; bi = p; }
  }
  nn[i] = bi;
}

// ------------------- edge relative-position features (E x 10) --------------
__global__ void k_build_rel(const float* __restrict__ pos, const int* __restrict__ col,
                            int E, float* __restrict__ rel) {
  int e = blockIdx.x * blockDim.x + threadIdx.x;
  if (e >= E) return;
  int i = col[e];          // aggregation index (target_to_source: i = col)
  int j = e >> 4;          // row = repeat(arange(m), 16)
  float ix = pos[3 * i + 0], iy = pos[3 * i + 1], iz = pos[3 * i + 2];
  float jx = pos[3 * j + 0], jy = pos[3 * j + 1], jz = pos[3 * j + 2];
  float dx = jx - ix, dy = jy - iy, dz = jz - iz;
  float d2 = fmaxf(dx * dx + dy * dy + dz * dz, 1e-12f);
  float* o = rel + (size_t)e * 10;
  o[0] = ix; o[1] = iy; o[2] = iz;
  o[3] = jx; o[4] = jy; o[5] = jz;
  o[6] = dx; o[7] = dy; o[8] = dz;
  o[9] = sqrtf(d2);
}

// ------------------- feats = concat(h[e>>4], lse[e]) -----------------------
__global__ void k_concat_rowgather(const float* __restrict__ A, const float* __restrict__ B,
                                   int E, int d1, int d2, float* __restrict__ out) {
  int t = blockIdx.x * blockDim.x + threadIdx.x;
  int d = d1 + d2;
  if (t >= E * d) return;
  int e = t / d, c = t % d;
  out[t] = (c < d1) ? A[(size_t)(e >> 4) * d1 + c]
                    : B[(size_t)e * d2 + (c - d1)];
}

// ------------------- out[r] = concat(A[idx[r]], B[r]) ----------------------
__global__ void k_concat_idxgather(const float* __restrict__ A, const float* __restrict__ B,
                                   const int* __restrict__ idx,
                                   int Ms, int d1, int d2, float* __restrict__ out) {
  int t = blockIdx.x * blockDim.x + threadIdx.x;
  int d = d1 + d2;
  if (t >= Ms * d) return;
  int r = t / d, c = t % d;
  out[t] = (c < d1) ? A[(size_t)idx[r] * d1 + c]
                    : B[(size_t)r * d2 + (c - d1)];
}

// ------------------- row softmax (one wave per row) ------------------------
__global__ __launch_bounds__(32) void k_softmax(float* __restrict__ S, int E, int d) {
  int r = blockIdx.x;
  if (r >= E) return;
  float* row = S + (size_t)r * d;
  int lane = threadIdx.x & 31;
  float mx = -3.4e38f;
  for (int c = lane; c < d; c += 32) mx = fmaxf(mx, row[c]);
  for (int o = 16; o > 0; o >>= 1) mx = fmaxf(mx, __shfl_xor(mx, o, 32));
  float s = 0.0f;
  for (int c = lane; c < d; c += 32) { float v = expf(row[c] - mx); row[c] = v; s += v; }
  for (int o = 16; o > 0; o >>= 1) s += __shfl_xor(s, o, 32);
  float inv = 1.0f / s;
  for (int c = lane; c < d; c += 32) row[c] *= inv;
}

// ------------------- scatter-add: agg[col[e]] += scores*feats --------------
__global__ void k_scatter(const float* __restrict__ scores, const float* __restrict__ feats,
                          const int* __restrict__ col, int E, int d,
                          float* __restrict__ agg) {
  int t = blockIdx.x * blockDim.x + threadIdx.x;
  if (t >= E * d) return;
  int e = t / d, c = t % d;
  atomicAdd(&agg[(size_t)col[e] * d + c], scores[t] * feats[t]);
}

__global__ void k_fill0(float* __restrict__ Y, int n) {
  int t = blockIdx.x * blockDim.x + threadIdx.x;
  if (t < n) Y[t] = 0.0f;
}

__global__ void k_copy(const float* __restrict__ X, float* __restrict__ Y, int n) {
  int t = blockIdx.x * blockDim.x + threadIdx.x;
  if (t < n) Y[t] = X[t];
}

// ------------------- a = lrelu(a + b) --------------------------------------
__global__ void k_add_lrelu(float* __restrict__ a, const float* __restrict__ b, int n) {
  int t = blockIdx.x * blockDim.x + threadIdx.x;
  if (t >= n) return;
  float v = a[t] + b[t];
  a[t] = (v >= 0.0f) ? v : LRELU_SLOPE * v;
}

// ------------------- Y[r] = X[4r]  (decimation gather) ---------------------
__global__ void k_decimate(const float* __restrict__ X, int mOut, int d,
                           float* __restrict__ Y) {
  int t = blockIdx.x * blockDim.x + threadIdx.x;
  if (t >= mOut * d) return;
  int r = t / d, c = t % d;
  Y[t] = X[(size_t)(4 * r) * d + c];
}

// ------------------- row log_softmax ---------------------------------------
__global__ void k_logsoftmax(const float* __restrict__ X, float* __restrict__ Y,
                             int M, int d) {
  int r = blockIdx.x * blockDim.x + threadIdx.x;
  if (r >= M) return;
  const float* in = X + (size_t)r * d;
  float mx = -3.4e38f;
  for (int c = 0; c < d; ++c) mx = fmaxf(mx, in[c]);
  float s = 0.0f;
  for (int c = 0; c < d; ++c) s += expf(in[c] - mx);
  float l = logf(s);
  for (int c = 0; c < d; ++c) Y[(size_t)r * d + c] = in[c] - mx - l;
}

// ===========================================================================
// Host orchestration
// ===========================================================================
struct MlpL { const float *W, *b, *g, *be; };
struct Blk  { MlpL mlp1, shortcut, mlp2;
              MlpL l1enc; const float* l1attn; MlpL l1post;
              MlpL l2enc; const float* l2attn; MlpL l2post; };

extern "C" void kernel_launch(void* const* d_in, const int* in_sizes, int n_in,
                              void* d_out, int out_size, void* d_ws, size_t ws_size,
                              hipStream_t stream) {
  (void)in_sizes; (void)n_in; (void)out_size; (void)ws_size;

  // ---- parameter walk (setup_inputs insertion order; leaves W,b,g,be) ----
  int cur = 0;
  auto NXT = [&]() -> const float* { return (const float*)d_in[cur++]; };
  auto rdM = [&]() { MlpL L; L.W = NXT(); L.b = NXT(); L.g = NXT(); L.be = NXT(); return L; };
  auto rdB = [&]() {
    Blk B;
    B.mlp1 = rdM(); B.shortcut = rdM(); B.mlp2 = rdM();
    B.l1enc = rdM(); B.l1attn = NXT(); B.l1post = rdM();
    B.l2enc = rdM(); B.l2attn = NXT(); B.l2post = rdM();
    return B;
  };

  const float* x   = NXT();
  const float* pos = NXT();
  const float* fc0W = NXT(); const float* fc0b = NXT();
  Blk b1 = rdB(), b2 = rdB(), b3 = rdB(), b4 = rdB();
  MlpL Lmlp1 = rdM();
  MlpL Lfp4 = rdM(), Lfp3 = rdM(), Lfp2 = rdM(), Lfp1 = rdM();
  MlpL Lhm0 = rdM(), Lhm1 = rdM();
  const float* hlW = NXT(); const float* hlb = NXT();

  // ---- workspace bump allocator (extra 256B tail slack per allocation so
  //      the GEMM's zero-weighted K-tail overread always stays in-bounds) ----
  char* wsb = (char*)d_ws;
  size_t wo = 0;
  auto allocf = [&](size_t elems) -> float* {
    float* p = (float*)(wsb + wo);
    wo += (((elems * sizeof(float)) + 255) & ~(size_t)255) + 256;
    return p;
  };
  auto alloci = [&](size_t elems) -> int* {
    int* p = (int*)(wsb + wo);
    wo += (((elems * sizeof(int)) + 255) & ~(size_t)255) + 256;
    return p;
  };

  float* stats = allocf(2048);   // mu @ stats, var @ stats+1024

  // GEMM: pack weights to f16 fragments, then unconditional WMMA kernel.
  // Pack scratch is reclaimed immediately (stream order keeps it safe).
  auto gemm = [&](const float* A, const float* W, const float* bias, float* C,
                  int M, int K, int N) {
    const int KC = (K + 31) / 32;
    const int TN = (N + 15) / 16;
    size_t gmark = wo;
    unsigned int* Wp = (unsigned int*)allocf((size_t)TN * KC * 32 * 8);
    k_packw<<<TN * KC, 32, 0, stream>>>(W, Wp, K, N, KC);
    dim3 g((unsigned)(M / 16), (unsigned)TN);   // M is always a multiple of 16
    k_gemm_f<<<g, 32, 0, stream>>>(A, Wp, bias, C, M, K, N, KC);
    wo = gmark;
  };

  auto linbn = [&](const float* X, int M, const MlpL& L, int din, int dout,
                   float* Y, bool act) {
    gemm(X, L.W, L.b, Y, M, din, dout);
    k_colstats<<<dout, 256, 0, stream>>>(Y, M, dout, stats, stats + 1024);
    int tot = M * dout;
    k_bn_act<<<(tot + 255) / 256, 256, 0, stream>>>(Y, stats, stats + 1024,
                                                    L.g, L.be, tot, dout, act ? 1 : 0);
  };

  // LFA: rel(E,10), h(n,dh) -> out(n,2*dh)
  auto lfa = [&](const int* col, const float* rel, const float* h, int n, int E,
                 int dh, const MlpL& enc, const float* attnW, const MlpL& post,
                 float* out) {
    int d = 2 * dh;
    float* lse = allocf((size_t)E * dh);
    linbn(rel, E, enc, 10, dh, lse, true);
    float* feats = allocf((size_t)E * d);
    k_concat_rowgather<<<(E * d + 255) / 256, 256, 0, stream>>>(h, lse, E, dh, dh, feats);
    float* scores = allocf((size_t)E * d);
    gemm(feats, attnW, nullptr, scores, E, d, d);
    k_softmax<<<E, 32, 0, stream>>>(scores, E, d);
    float* agg = allocf((size_t)n * d);
    k_fill0<<<(n * d + 255) / 256, 256, 0, stream>>>(agg, n * d);
    k_scatter<<<(E * d + 255) / 256, 256, 0, stream>>>(scores, feats, col, E, d, agg);
    linbn(agg, n, post, d, d, out, true);
  };

  // Encoder block: (x(n,din), pos(n,3)) -> (xout(n/4,dout), posout(n/4,3))
  auto block = [&](const float* xin, const float* posin, int n, int din, int dout,
                   const Blk& B, float* xout, float* posout) {
    size_t mark = wo;
    int m = n / 4, E = m * 16;
    float* posQ = allocf((size_t)m * 3);
    k_decimate<<<(m * 3 + 255) / 256, 256, 0, stream>>>(posin, m, 3, posQ);
    int* col = alloci((size_t)E);
    k_knn<<<(m + 63) / 64, 64, 0, stream>>>(posQ, posin, m, n, col);
    float* rel = allocf((size_t)E * 10);
    k_build_rel<<<(E + 255) / 256, 256, 0, stream>>>(posin, col, E, rel);

    float* sc = allocf((size_t)n * dout);
    linbn(xin, n, B.shortcut, din, dout, sc, false);
    float* h1 = allocf((size_t)n * (dout / 8));
    linbn(xin, n, B.mlp1, din, dout / 8, h1, true);
    float* h2 = allocf((size_t)n * (dout / 4));
    lfa(col, rel, h1, n, E, dout / 8, B.l1enc, B.l1attn, B.l1post, h2);
    float* h3 = allocf((size_t)n * (dout / 2));
    lfa(col, rel, h2, n, E, dout / 4, B.l2enc, B.l2attn, B.l2post, h3);
    float* h4 = allocf((size_t)n * dout);
    linbn(h3, n, B.mlp2, dout / 2, dout, h4, false);
    k_add_lrelu<<<(n * dout + 255) / 256, 256, 0, stream>>>(h4, sc, n * dout);
    k_decimate<<<(m * dout + 255) / 256, 256, 0, stream>>>(h4, m, dout, xout);
    k_decimate<<<(m * 3 + 255) / 256, 256, 0, stream>>>(posin, m, 3, posout);
    wo = mark;   // stream-ordered execution makes scratch reuse safe
  };

  // Decoder FP: out(Ms,dout) = mlp(concat(xsrc[nn1], xskip))
  auto fp = [&](const float* xsrc, const float* possrc, int Mx, int dx,
                const float* xskip, const float* posskip, int Ms, int dskip,
                const MlpL& L, int dout, float* out) {
    size_t mark = wo;
    int* nn = alloci((size_t)Ms);
    k_nn1<<<(Ms + 63) / 64, 64, 0, stream>>>(posskip, Ms, possrc, Mx, nn);
    int dc = dx + dskip;
    float* hcat = allocf((size_t)Ms * dc);
    k_concat_idxgather<<<(Ms * dc + 255) / 256, 256, 0, stream>>>(xsrc, xskip, nn,
                                                                  Ms, dx, dskip, hcat);
    linbn(hcat, Ms, L, dc, dout, out, true);
    wo = mark;
  };

  // ---- persistent activations ----
  const int N = 16384, BNECK = 9;
  float* xw = allocf((size_t)N * BNECK);   // copy of input x (safe A operand)
  float* h0 = allocf((size_t)N * BNECK);
  float* x1 = allocf((size_t)4096 * 32);  float* p1 = allocf((size_t)4096 * 3);
  float* x2 = allocf((size_t)1024 * 128); float* p2 = allocf((size_t)1024 * 3);
  float* x3 = allocf((size_t)256 * 256);  float* p3 = allocf((size_t)256 * 3);
  float* x4 = allocf((size_t)64 * 512);   float* p4 = allocf((size_t)64 * 3);
  float* hd  = allocf((size_t)64 * 512);
  float* hf4 = allocf((size_t)256 * 256);
  float* hf3 = allocf((size_t)1024 * 128);
  float* hf2 = allocf((size_t)4096 * 32);
  float* hf1 = allocf((size_t)N * BNECK);
  float* hh0 = allocf((size_t)N * 64);
  float* hh1 = allocf((size_t)N * 32);
  float* logits = allocf((size_t)N * 7);

  // ---- forward pass ----
  k_copy<<<(N * BNECK + 255) / 256, 256, 0, stream>>>(x, xw, N * BNECK);
  gemm(xw, fc0W, fc0b, h0, N, 9, BNECK);                        // fc0
  block(h0, pos, N,    BNECK, 32,  b1, x1, p1);
  block(x1, p1, 4096,  32,   128,  b2, x2, p2);
  block(x2, p2, 1024,  128,  256,  b3, x3, p3);
  block(x3, p3, 256,   256,  512,  b4, x4, p4);
  linbn(x4, 64, Lmlp1, 512, 512, hd, true);                     // mlp1
  fp(hd,  p4, 64,   512, x3, p3, 256,  256, Lfp4, 256, hf4);
  fp(hf4, p3, 256,  256, x2, p2, 1024, 128, Lfp3, 128, hf3);
  fp(hf3, p2, 1024, 128, x1, p1, 4096, 32,  Lfp2, 32,  hf2);
  fp(hf2, p1, 4096, 32,  h0, pos, N,   BNECK, Lfp1, BNECK, hf1);
  linbn(hf1, N, Lhm0, BNECK, 64, hh0, true);                    // head mlp L0
  linbn(hh0, N, Lhm1, 64, 32, hh1, true);                       // head mlp L1
  gemm(hh1, hlW, hlb, logits, N, 32, 7);                        // head linear
  k_logsoftmax<<<(N + 63) / 64, 64, 0, stream>>>(logits, (float*)d_out, N, 7);
}
